// Expert_76132590288998
// MI455X (gfx1250) — compile-verified, matrix operand order checked by reading, not run on hardware
//
#include <hip/hip_runtime.h>
#include <hip/hip_bf16.h>
#include <math.h>

// ---------------------------------------------------------------------------
// Types for CDNA5 WMMA
// ---------------------------------------------------------------------------
typedef __attribute__((ext_vector_type(16))) __bf16 bf16x16;
typedef __attribute__((ext_vector_type(8)))  __bf16 bf16x8;
typedef __attribute__((ext_vector_type(8)))  float  v8f;

#define E_EXP 8
#define BM 128
#define BN 64
#define BK 64
#define LDT 72                       // padded LDS row stride in halves (+8 halves)
#define A_HALVES (BM * LDT)          // 9216
#define B_HALVES (BN * LDT)          // 4608
#define BUF_HALVES (A_HALVES + B_HALVES) // 13824

// round-to-nearest-even f32 -> bf16 (raw u16)
static __device__ __forceinline__ unsigned int f32_to_bf16_bits(float f) {
    unsigned int u = __float_as_uint(f);
    unsigned int r = u + 0x7FFFu + ((u >> 16) & 1u);
    return r >> 16;
}
static __device__ __forceinline__ unsigned int pack_bf16x2(float lo, float hi) {
    return (f32_to_bf16_bits(hi) << 16) | f32_to_bf16_bits(lo);
}

// ---------------------------------------------------------------------------
// CDNA5 async global->LDS copy (ASYNCcnt-tracked) + wait
// Generic __shared__ pointers: low 32 bits == LDS byte address (aperture rule).
// ---------------------------------------------------------------------------
static __device__ __forceinline__ void async_copy_b128(const void* gptr, void* lptr) {
    unsigned int       laddr = (unsigned int)(size_t)lptr;
    unsigned long long gaddr = (unsigned long long)(size_t)gptr;
    asm volatile("global_load_async_to_lds_b128 %0, %1, off"
                 :: "v"(laddr), "v"(gaddr) : "memory");
}
static __device__ __forceinline__ void wait_asynccnt0() {
    asm volatile("s_wait_asynccnt 0" ::: "memory");
}

// ---------------------------------------------------------------------------
// Weight pre-conversion: f32 -> bf16 (vectorized x4)
// ---------------------------------------------------------------------------
__global__ __launch_bounds__(256) void cvt_f32_to_bf16(const float4* __restrict__ in,
                                                       uint2* __restrict__ out, int n4) {
    int i = blockIdx.x * blockDim.x + threadIdx.x;
    if (i < n4) {
        float4 v = in[i];
        uint2 o;
        o.x = pack_bf16x2(v.x, v.y);
        o.y = pack_bf16x2(v.z, v.w);
        out[i] = o;
    }
}

// ---------------------------------------------------------------------------
// LDS fragment load (A 16x32 / B 32x16 bf16, transpose-symmetric layout):
// lane l: row = l&15, halves [kb0..kb0+7] and [kb0+16..kb0+23],
// where kb0 already includes (l>>4)*8.
// ---------------------------------------------------------------------------
static __device__ __forceinline__ bf16x16 ld_frag(const unsigned short* p) {
    bf16x8 lo = *(const bf16x8*)(p);
    bf16x8 hi = *(const bf16x8*)(p + 16);
    bf16x16 r;
#pragma unroll
    for (int i = 0; i < 8; ++i) { r[i] = lo[i]; r[i + 8] = hi[i]; }
    return r;
}

// A staging: scale by wb[row,e], convert to bf16, keep in regs
template<int I>
static __device__ __forceinline__ void load_a_regs(uint4 (&rA)[4], const float* __restrict__ X,
                                                   const float* __restrict__ wbp,
                                                   int blockM, int e, int ibase,
                                                   int ldrow, int ldcol) {
#pragma unroll
    for (int p = 0; p < 4; ++p) {
        const int grow = blockM + p * 32 + ldrow;
        const float s = wbp[grow * E_EXP + e];
        const float4* src = (const float4*)(X + (size_t)grow * I + ibase + ldcol);
        float4 a = src[0], b = src[1];
        rA[p].x = pack_bf16x2(a.x * s, a.y * s);
        rA[p].y = pack_bf16x2(a.z * s, a.w * s);
        rA[p].z = pack_bf16x2(b.x * s, b.y * s);
        rA[p].w = pack_bf16x2(b.z * s, b.w * s);
    }
}

static __device__ __forceinline__ void store_a_lds(unsigned short* As, const uint4 (&rA)[4],
                                                   int ldrow, int ldcol) {
#pragma unroll
    for (int p = 0; p < 4; ++p)
        *(uint4*)(As + (p * 32 + ldrow) * LDT + ldcol) = rA[p];
}

// B staging: async copy bf16 weights directly into LDS (2 x b128 per thread)
template<int I, int O>
static __device__ __forceinline__ void async_load_b(const unsigned short* __restrict__ Wb,
                                                    unsigned short* Bs, int blockN,
                                                    int e, int ibase, int ldrow, int ldcol) {
#pragma unroll
    for (int p = 0; p < 2; ++p) {
        const int row = p * 32 + ldrow;
        const unsigned short* g =
            Wb + (size_t)e * O * I + (size_t)(blockN + row) * I + ibase + ldcol;
        async_copy_b128(g, Bs + row * LDT + ldcol);
    }
}

static __device__ __forceinline__ void compute_block(const unsigned short* As,
                                                     const unsigned short* Bs,
                                                     v8f (&acc)[2][2],
                                                     int waveM, int waveN,
                                                     int lrow, int khalf) {
#pragma unroll
    for (int ks = 0; ks < 2; ++ks) {
        const int kb0 = ks * 32 + khalf;
        bf16x16 bfrag0 = ld_frag(Bs + (waveN * 32 +  0 + lrow) * LDT + kb0);
        bf16x16 bfrag1 = ld_frag(Bs + (waveN * 32 + 16 + lrow) * LDT + kb0);
#pragma unroll
        for (int mi = 0; mi < 2; ++mi) {
            bf16x16 afrag = ld_frag(As + (waveM * 32 + mi * 16 + lrow) * LDT + kb0);
            acc[mi][0] = __builtin_amdgcn_wmma_f32_16x16x32_bf16(
                false, afrag, false, bfrag0, (short)0, acc[mi][0], false, false);
            acc[mi][1] = __builtin_amdgcn_wmma_f32_16x16x32_bf16(
                false, afrag, false, bfrag1, (short)0, acc[mi][1], false, false);
        }
    }
}

// ---------------------------------------------------------------------------
// Fused blended-expert GEMM layer:
//   Y[b,o] = act( sum_{e,i} (wb[b,e]*X[b,i]) * W[e,o,i] + sum_e wb[b,e]*Bias[e,o] )
// One flat GEMM with K = E*I; A built on the fly (wb-scale fused), W bf16.
// ---------------------------------------------------------------------------
template<int I, int O>
__global__ __launch_bounds__(256) void moe_gemm(const float* __restrict__ X,          // [2048, I]
                                                const unsigned short* __restrict__ Wb,// [E, O, I] bf16
                                                const float* __restrict__ wbp,        // [2048, E]
                                                const float* __restrict__ Bias,       // [E, O]
                                                float* __restrict__ Y,                // [2048, O]
                                                int act) {
    constexpr int LOG2I = (I == 512) ? 9 : 10;
    constexpr int NKB   = (E_EXP * I) >> 6;   // K blocks of 64

    __shared__ __align__(16) unsigned short smem[2 * BUF_HALVES];

    const int tid    = threadIdx.x;
    const int lane   = tid & 31;
    const int wave   = tid >> 5;
    const int waveM  = wave & 3;          // 0..3 : 32 rows each
    const int waveN  = wave >> 2;         // 0..1 : 32 cols each
    const int lrow   = lane & 15;
    const int khalf  = (lane >> 4) << 3;  // 0 or 8

    const int blockN = blockIdx.x * BN;
    const int blockM = blockIdx.y * BM;

    const int ldcol  = (tid & 7) * 8;     // staging column (halves / floats)
    const int ldrow  = tid >> 3;          // staging row within a 32-row pass

    v8f acc[2][2];
#pragma unroll
    for (int mi = 0; mi < 2; ++mi)
#pragma unroll
        for (int ni = 0; ni < 2; ++ni)
#pragma unroll
            for (int j = 0; j < 8; ++j) acc[mi][ni][j] = 0.0f;

    uint4 rA[4];

    // ---- prologue: stage K-block 0 (A -> regs, B -> async LDS) ----
    load_a_regs<I>(rA, X, wbp, blockM, 0, 0, ldrow, ldcol);
    async_load_b<I, O>(Wb, smem + A_HALVES, blockN, 0, 0, ldrow, ldcol);

    int buf = 0;
    // ---- main loop (branch-free body; last block peeled) ----
    for (int kb = 0; kb < NKB - 1; ++kb) {
        unsigned short* As = smem + buf * BUF_HALVES;
        unsigned short* Bs = As + A_HALVES;

        store_a_lds(As, rA, ldrow, ldcol);
        wait_asynccnt0();                  // B tile of this buffer complete
        __syncthreads();

        // prefetch next K-block into the other buffer / regs
        const int k64   = (kb + 1) << 6;
        const int e     = k64 >> LOG2I;
        const int ibase = k64 & (I - 1);
        unsigned short* Asn = smem + (buf ^ 1) * BUF_HALVES;
        async_load_b<I, O>(Wb, Asn + A_HALVES, blockN, e, ibase, ldrow, ldcol);
        load_a_regs<I>(rA, X, wbp, blockM, e, ibase, ldrow, ldcol);

        compute_block(As, Bs, acc, waveM, waveN, lrow, khalf);
        buf ^= 1;
    }
    // ---- peeled last K-block ----
    {
        unsigned short* As = smem + buf * BUF_HALVES;
        unsigned short* Bs = As + A_HALVES;
        store_a_lds(As, rA, ldrow, ldcol);
        wait_asynccnt0();
        __syncthreads();
        compute_block(As, Bs, acc, waveM, waveN, lrow, khalf);
    }

    // ---- epilogue: blended bias + optional ELU, store f32 ----
    const int m0 = blockM + waveM * 32;
    const int n0 = blockN + waveN * 32;

    float bias_e[2][E_EXP];
#pragma unroll
    for (int ni = 0; ni < 2; ++ni) {
        const int col = n0 + ni * 16 + lrow;
#pragma unroll
        for (int e = 0; e < E_EXP; ++e) bias_e[ni][e] = Bias[e * O + col];
    }

#pragma unroll
    for (int mi = 0; mi < 2; ++mi) {
#pragma unroll
        for (int r = 0; r < 8; ++r) {
            const int row = m0 + mi * 16 + r + khalf;
            float wbr[E_EXP];
#pragma unroll
            for (int e = 0; e < E_EXP; ++e) wbr[e] = wbp[row * E_EXP + e];
#pragma unroll
            for (int ni = 0; ni < 2; ++ni) {
                const int col = n0 + ni * 16 + lrow;
                float v = acc[mi][ni][r];
                float bb = 0.0f;
#pragma unroll
                for (int e = 0; e < E_EXP; ++e) bb += wbr[e] * bias_e[ni][e];
                v += bb;
                if (act) v = (v > 0.0f) ? v : expm1f(v);
                Y[(size_t)row * O + col] = v;
            }
        }
    }
}

// ---------------------------------------------------------------------------
// Host-side launcher
// ---------------------------------------------------------------------------
extern "C" void kernel_launch(void* const* d_in, const int* in_sizes, int n_in,
                              void* d_out, int out_size, void* d_ws, size_t ws_size,
                              hipStream_t stream) {
    const float* wb = (const float*)d_in[0];  // [2048, 8]
    const float* x  = (const float*)d_in[1];  // [2048, 512]
    const float* W0 = (const float*)d_in[2];  // [8, 1024, 512]
    const float* B0 = (const float*)d_in[3];  // [8, 1024]
    const float* W1 = (const float*)d_in[4];  // [8, 1024, 1024]
    const float* B1 = (const float*)d_in[5];  // [8, 1024]
    const float* W2 = (const float*)d_in[6];  // [8, 512, 1024]
    const float* B2 = (const float*)d_in[7];  // [8, 512]

    char* ws = (char*)d_ws;
    unsigned short* W0b = (unsigned short*)(ws);               // 8 MiB
    unsigned short* W1b = (unsigned short*)(ws + (8u << 20));  // 16 MiB
    unsigned short* W2b = (unsigned short*)(ws + (24u << 20)); // 8 MiB
    float* h1 = (float*)(ws + (32u << 20));                    // 8 MiB
    float* h2 = (float*)(ws + (40u << 20));                    // 8 MiB

    // 1) convert weights f32 -> bf16 (per-call, deterministic)
    const int n4_0 = (8 * 1024 * 512) / 4;
    const int n4_1 = (8 * 1024 * 1024) / 4;
    const int n4_2 = (8 * 512 * 1024) / 4;
    cvt_f32_to_bf16<<<(n4_0 + 255) / 256, 256, 0, stream>>>((const float4*)W0, (uint2*)W0b, n4_0);
    cvt_f32_to_bf16<<<(n4_1 + 255) / 256, 256, 0, stream>>>((const float4*)W1, (uint2*)W1b, n4_1);
    cvt_f32_to_bf16<<<(n4_2 + 255) / 256, 256, 0, stream>>>((const float4*)W2, (uint2*)W2b, n4_2);

    // 2) three fused layers (grid = (O/BN, M/BM))
    moe_gemm< 512, 1024><<<dim3(1024 / BN, 2048 / BM), 256, 0, stream>>>(x,  W0b, wb, B0, h1, 1);
    moe_gemm<1024, 1024><<<dim3(1024 / BN, 2048 / BM), 256, 0, stream>>>(h1, W1b, wb, B1, h2, 1);
    moe_gemm<1024,  512><<<dim3(512  / BN, 2048 / BM), 256, 0, stream>>>(h2, W2b, wb, B2, (float*)d_out, 0);
}
